// CoAttention_55293408969066
// MI455X (gfx1250) — matrix operand order repaired and checked by reading, not compiled, not run on hardware
//
#include <hip/hip_runtime.h>
#include <hip/hip_bf16.h>
#include <stdint.h>

typedef __bf16 bf16;
typedef __attribute__((ext_vector_type(16))) __bf16 v16bf;
typedef __attribute__((ext_vector_type(8)))  float  v8f;
typedef uint32_t su4 __attribute__((ext_vector_type(4)));
typedef uint32_t su8 __attribute__((ext_vector_type(8)));

union ABu { uint4 u[2]; v16bf v; };

constexpr int CB  = 64;    // batch
constexpr int CNV = 2048;  // Nv
constexpr int CNQ = 512;   // Nq
constexpr int CH  = 256;   // H
constexpr int CHA = 128;   // Ha

// LDS byte address of a flat pointer to __shared__ data: low 32 bits of the
// flat address are the LDS offset (ISA 10.2, LDS aperture).
__device__ __forceinline__ uint32_t lds_addr_of(const void* p) {
  return (uint32_t)(uintptr_t)p;
}

// Async global -> LDS copy of 16 bytes (CDNA5 GLOBAL_LOAD_ASYNC_TO_LDS_B128, ASYNCcnt).
__device__ __forceinline__ void async_copy_b128(uint32_t lds, const void* gaddr) {
  asm volatile("global_load_async_to_lds_b128 %0, %1, off"
               :: "v"(lds), "v"((uint64_t)(uintptr_t)gaddr)
               : "memory");
}
__device__ __forceinline__ void async_wait_all() {
  asm volatile("s_wait_asynccnt 0" ::: "memory");
}

// Tensor Data Mover: one 2D tile (rows x 64B) global -> LDS, with LDS padding
// of 8 dwords after every 16 dwords (reproduces the 96B LDS row pitch).
// D# layout per CDNA5 ISA 8.3/8.4: group0 = {count, lds_addr, global_addr, type=2},
// group1 = {data_size=4B, pad_enable, pad_interval=16dw, pad_amount=8dw,
//           tensor_dim0/stride = row_stride_dw, tile_dim0=16dw, tile_dim1=rows}.
__device__ __forceinline__ void tdm_load_tile_2d(uint32_t lds, const void* gaddr,
                                                 uint32_t row_stride_dw, uint32_t rows) {
  const uint64_t ga = (uint64_t)(uintptr_t)gaddr;
  const su4 g0 = {
    1u,                                                   // count=1 (user D#)
    lds,                                                  // lds_addr
    (uint32_t)ga,                                         // global_addr[31:0]
    ((uint32_t)(ga >> 32) & 0x01FFFFFFu) | (2u << 30)     // global_addr[56:32] | type=2
  };
  const uint32_t tdim1 = 1u << 20;                        // tensor rows (never OOB here)
  const su8 g1 = {
    (2u << 16) | (1u << 20) | (3u << 22) | (7u << 25),    // data_size=4B, pad en, 16dw/8dw
    (row_stride_dw & 0xFFFFu) << 16,                      // tensor_dim0[15:0]
    (row_stride_dw >> 16) | ((tdim1 & 0xFFFFu) << 16),    // tensor_dim0[31:16] | td1[15:0]
    (tdim1 >> 16) | (16u << 16),                          // td1[31:16] | tile_dim0=16 dw
    rows & 0xFFFFu,                                       // tile_dim1 (tile_dim2=0)
    row_stride_dw,                                        // tensor_dim0_stride[31:0]
    0u,                                                   // stride hi | dim1_stride lo
    0u
  };
  asm volatile("tensor_load_to_lds %0, %1" :: "s"(g0), "s"(g1) : "memory");
}

// ---------------- cast fp32 -> bf16, 8 elems/thread ----------------
__global__ __launch_bounds__(256)
void cast_f32_to_bf16(const float* __restrict__ x, bf16* __restrict__ y, long long n8) {
  long long i = (long long)blockIdx.x * 256 + threadIdx.x;
  if (i >= n8) return;
  const float4* xs = (const float4*)x;
  float4 f0 = xs[2 * i], f1 = xs[2 * i + 1];
  union { bf16 h[8]; uint4 u; } p;
  p.h[0] = (bf16)f0.x; p.h[1] = (bf16)f0.y; p.h[2] = (bf16)f0.z; p.h[3] = (bf16)f0.w;
  p.h[4] = (bf16)f1.x; p.h[5] = (bf16)f1.y; p.h[6] = (bf16)f1.z; p.h[7] = (bf16)f1.w;
  ((uint4*)y)[i] = p.u;
}

// ---------------- generic NT GEMM with fused epilogue ----------------
// D = act( A[M,K] @ Bw[N,K]^T + bias + Add ), optionally fused dot with wdot -> sOut[m]
// Block tile 64(M) x 128(N), 8 wave32 waves, each wave 32x32 via 4 WMMA accumulators.
// Double-buffered LDS: A tile via per-lane async global->LDS copies, B tile via one
// TDM descriptor per chunk (TENSORcnt), both with one-chunk lookahead.
// biasMode: 0=none, 1=bias[n], 2=bias[m].  act: 0=none, 1=tanh.
__global__ __launch_bounds__(256)
void gemm_nt_kernel(const bf16* __restrict__ A, int lda, long long sA,
                    const bf16* __restrict__ Bw, int ldb, long long sB,
                    bf16* __restrict__ C, int ldc, long long sC,
                    const float* __restrict__ bias, int biasMode,
                    const bf16* __restrict__ Add, int ldAdd, long long sAdd,
                    const float* __restrict__ wdot, float* __restrict__ sOut, long long sS,
                    int K, int act)
{
  // LDS tiles: rows padded to 48 bf16 (96B = 6 uint4) -> 16B-aligned b128 gathers.
  __shared__ uint4 As[2][64 * 6];
  __shared__ uint4 Bs[2][128 * 6];
  __shared__ float sdot4[4][64];

  const int zb = blockIdx.z;
  A  += (long long)zb * sA;
  Bw += (long long)zb * sB;
  if (C)    C    += (long long)zb * sC;
  if (Add)  Add  += (long long)zb * sAdd;
  if (sOut) sOut += (long long)zb * sS;

  const int t    = threadIdx.x;
  const int m0   = blockIdx.y * 64;
  const int n0   = blockIdx.x * 128;
  const int lane = t & 31;          // wave32
  const int w    = t >> 5;          // wave id 0..7
  const int wm   = w & 1;           // 2 waves along M
  const int wn   = w >> 1;          // 4 waves along N
  const int hf   = lane >> 4;       // half-wave (ISA A/B K-split)
  const int lid  = lane & 15;       // row / column within 16

  const v8f zero = {0.f, 0.f, 0.f, 0.f, 0.f, 0.f, 0.f, 0.f};
  v8f acc[2][2];
  acc[0][0] = zero; acc[0][1] = zero; acc[1][0] = zero; acc[1][1] = zero;

  const int arow = t >> 2, aseg = t & 3;     // A stage: 256 x b128

  // Stage one 32-K chunk into LDS buffer `buf`.
  auto stage_chunk = [&](int k0, int buf) {
    const bf16* sa = A + (size_t)(m0 + arow) * lda + (size_t)k0 + aseg * 8;
    async_copy_b128(lds_addr_of(&As[buf][arow * 6 + aseg]), sa);
    if (w == 0) {  // one TDM descriptor moves the whole 128x64B B tile
      const bf16* sb = Bw + (size_t)n0 * ldb + (size_t)k0;
      tdm_load_tile_2d(lds_addr_of(&Bs[buf][0]), sb, (uint32_t)(ldb >> 1), 128u);
    }
  };

  const int nchunks = K >> 5;   // K / 32, K is always a multiple of 32 here

  // Prologue: fill buffer 0.
  stage_chunk(0, 0);
  async_wait_all();
  __builtin_amdgcn_s_wait_tensorcnt(0);
  __syncthreads();

  for (int i = 0; i < nchunks; ++i) {
    const int buf = i & 1;
    if (i + 1 < nchunks) stage_chunk((i + 1) << 5, buf ^ 1);   // async/TDM lookahead

    // Gather WMMA operands per the CDNA5 16-bit A (16x32) / B (32x16) VGPR layouts.
    // A lane (hf,lid): K runs [8*hf, 8*hf+8) and [16+8*hf, 16+8*hf+8)  -> uint4 {hf, 2+hf}
    // B lane (hf,lid): K runs [8*hf, 8*hf+8) and [16+8*hf, 16+8*hf+8)  -> uint4 {hf, 2+hf}
    ABu a0, a1, b0, b1;
    const int ra = (wm * 32 + lid) * 6;
    a0.u[0] = As[buf][ra + hf];           a0.u[1] = As[buf][ra + 2 + hf];
    a1.u[0] = As[buf][ra + 96 + hf];      a1.u[1] = As[buf][ra + 96 + 2 + hf];
    const int rb = (wn * 32 + lid) * 6;
    b0.u[0] = Bs[buf][rb + hf];           b0.u[1] = Bs[buf][rb + 2 + hf];
    b1.u[0] = Bs[buf][rb + 96 + hf];      b1.u[1] = Bs[buf][rb + 96 + 2 + hf];

    acc[0][0] = __builtin_amdgcn_wmma_f32_16x16x32_bf16(false, a0.v, false, b0.v, (short)0, acc[0][0], false, false);
    acc[0][1] = __builtin_amdgcn_wmma_f32_16x16x32_bf16(false, a0.v, false, b1.v, (short)0, acc[0][1], false, false);
    acc[1][0] = __builtin_amdgcn_wmma_f32_16x16x32_bf16(false, a1.v, false, b0.v, (short)0, acc[1][0], false, false);
    acc[1][1] = __builtin_amdgcn_wmma_f32_16x16x32_bf16(false, a1.v, false, b1.v, (short)0, acc[1][1], false, false);

    // Lookahead writes (buf^1) must have landed before anyone proceeds;
    // this wave's reads of buf completed above.
    async_wait_all();
    __builtin_amdgcn_s_wait_tensorcnt(0);
    __syncthreads();
  }

  // ---- epilogue: C/D f32 layout -> element r of lane (hf,lid): M = r + 8*hf, N = lid
#pragma unroll
  for (int mi = 0; mi < 2; ++mi) {
    float pd[8] = {0, 0, 0, 0, 0, 0, 0, 0};
#pragma unroll
    for (int ni = 0; ni < 2; ++ni) {
      const int n = n0 + wn * 32 + ni * 16 + lid;
      const float bn = (biasMode == 1) ? bias[n] : 0.0f;
#pragma unroll
      for (int r = 0; r < 8; ++r) {
        const int m = m0 + wm * 32 + mi * 16 + r + 8 * hf;
        float val = acc[mi][ni][r] + bn;
        if (biasMode == 2) val += bias[m];
        if (Add)  val += (float)Add[(size_t)m * ldAdd + n];
        if (act)  val = tanhf(val);
        if (C)    C[(size_t)m * ldc + n] = (bf16)val;
        if (wdot) pd[r] += val * wdot[n];
      }
    }
    if (wdot) { // reduce across lid (16-lane groups), deterministic LDS combine across wn
#pragma unroll
      for (int r = 0; r < 8; ++r) {
        float x = pd[r];
        x += __shfl_xor(x, 1, 32);
        x += __shfl_xor(x, 2, 32);
        x += __shfl_xor(x, 4, 32);
        x += __shfl_xor(x, 8, 32);
        if (lid == 0) sdot4[wn][wm * 32 + mi * 16 + r + 8 * hf] = x;
      }
    }
  }
  if (sOut) {
    __syncthreads();
    if (t < 64) sOut[m0 + t] = sdot4[0][t] + sdot4[1][t] + sdot4[2][t] + sdot4[3][t];
  }
}

// ---------------- per-batch softmax over sequence + weighted sum ----------------
// out[b, :] = sum_n softmax(s[b,:])[n] * x[b, n, :]   (H = 256 = blockDim)
__global__ __launch_bounds__(256)
void softmax_weighted_sum(const float* __restrict__ s, const float* __restrict__ x,
                          float* __restrict__ out, int L) {
  __shared__ float red[256];
  __shared__ float pbuf[2048];
  const int b = blockIdx.x;
  const int t = threadIdx.x;
  s   += (size_t)b * L;
  x   += (size_t)b * L * CH;
  out += (size_t)b * CH;

  float mx = -3.4e38f;
  for (int i = t; i < L; i += 256) mx = fmaxf(mx, s[i]);
  red[t] = mx; __syncthreads();
  for (int k2 = 128; k2 > 0; k2 >>= 1) { if (t < k2) red[t] = fmaxf(red[t], red[t + k2]); __syncthreads(); }
  mx = red[0];
  __syncthreads();
  float se = 0.f;
  for (int i = t; i < L; i += 256) { float e = __expf(s[i] - mx); pbuf[i] = e; se += e; }
  red[t] = se; __syncthreads();
  for (int k2 = 128; k2 > 0; k2 >>= 1) { if (t < k2) red[t] += red[t + k2]; __syncthreads(); }
  const float inv = 1.0f / red[0];
  __syncthreads();
  float accum = 0.f;
  for (int n = 0; n < L; ++n) accum += pbuf[n] * x[(size_t)n * CH + t];
  out[t] = accum * inv;
}

// ---------------- host-side launch helper ----------------
static inline void launch_gemm(hipStream_t s, int M, int N, int K, int batch,
                               const bf16* A, int lda, long long sA,
                               const bf16* Bw, int ldb, long long sB,
                               bf16* C, int ldc, long long sC,
                               const float* bias, int biasMode, int act,
                               const bf16* Add, int ldAdd, long long sAdd,
                               const float* wdot, float* sOut, long long sS) {
  dim3 g(N / 128, M / 64, batch);
  gemm_nt_kernel<<<g, 256, 0, s>>>(A, lda, sA, Bw, ldb, sB, C, ldc, sC,
                                   bias, biasMode, Add, ldAdd, sAdd,
                                   wdot, sOut, sS, K, act);
}

extern "C" void kernel_launch(void* const* d_in, const int* in_sizes, int n_in,
                              void* d_out, int out_size, void* d_ws, size_t ws_size,
                              hipStream_t stream) {
  (void)in_sizes; (void)n_in; (void)out_size; (void)ws_size;

  const float* v    = (const float*)d_in[0];
  const float* q    = (const float*)d_in[1];
  const float* Waff = (const float*)d_in[2];
  const float* baff = (const float*)d_in[3];
  const float* Wv   = (const float*)d_in[4];
  const float* bv   = (const float*)d_in[5];
  const float* Wq   = (const float*)d_in[6];
  const float* bq   = (const float*)d_in[7];
  const float* Whv  = (const float*)d_in[8];
  const float* Whq  = (const float*)d_in[10];
  // bhv (d_in[9]) / bhq (d_in[11]) are added before softmax -> shift-invariant, omitted.

  uint8_t* wsp = (uint8_t*)d_ws;
  auto alloc = [&](size_t bytes) -> void* {
    void* p = (void*)wsp;
    wsp += (bytes + 255) & ~(size_t)255;
    return p;
  };

  bf16* vb    = (bf16*)alloc((size_t)CB * CNV * CH * 2);   // v in bf16
  bf16* qb    = (bf16*)alloc((size_t)CB * CNQ * CH * 2);   // q in bf16
  bf16* Waffb = (bf16*)alloc((size_t)CH * CH * 2);
  bf16* Wvb   = (bf16*)alloc((size_t)CHA * CH * 2);
  bf16* Wqb   = (bf16*)alloc((size_t)CHA * CH * 2);
  bf16* vwb   = (bf16*)alloc((size_t)CB * CNV * CH * 2);   // v @ Waff^T + baff
  bf16* pvb   = (bf16*)alloc((size_t)CB * CNV * CHA * 2);  // pv   [B, Nv, Ha]
  bf16* pvTb  = (bf16*)alloc((size_t)CB * CHA * CNV * 2);  // pv^T [B, Ha, Nv]
  bf16* pqb   = (bf16*)alloc((size_t)CB * CNQ * CHA * 2);  // pq   [B, Nq, Ha]
  bf16* pqTb  = (bf16*)alloc((size_t)CB * CHA * CNQ * 2);  // pq^T [B, Ha, Nq]
  bf16* affb  = (bf16*)alloc((size_t)CB * CNQ * CNV * 2);  // aff   [B, Nq, Nv]
  bf16* affTb = (bf16*)alloc((size_t)CB * CNV * CNQ * 2);  // aff^T [B, Nv, Nq]
  float* sv   = (float*)alloc((size_t)CB * CNV * 4);       // h_v . Whv
  float* sq   = (float*)alloc((size_t)CB * CNQ * 4);       // h_q . Whq

  // ---- 1) casts to bf16
  {
    long long n8;
    n8 = (long long)CB * CNV * CH / 8;
    cast_f32_to_bf16<<<dim3((unsigned)(n8 / 256)), 256, 0, stream>>>(v, vb, n8);
    n8 = (long long)CB * CNQ * CH / 8;
    cast_f32_to_bf16<<<dim3((unsigned)(n8 / 256)), 256, 0, stream>>>(q, qb, n8);
    n8 = (long long)CH * CH / 8;
    cast_f32_to_bf16<<<dim3((unsigned)(n8 / 256)), 256, 0, stream>>>(Waff, Waffb, n8);
    n8 = (long long)CHA * CH / 8;
    cast_f32_to_bf16<<<dim3((unsigned)(n8 / 256)), 256, 0, stream>>>(Wv, Wvb, n8);
    cast_f32_to_bf16<<<dim3((unsigned)(n8 / 256)), 256, 0, stream>>>(Wq, Wqb, n8);
  }

  const long long strV  = (long long)CNV * CH;   // v / vw batch stride
  const long long strQ  = (long long)CNQ * CH;
  const long long strPV = (long long)CNV * CHA;  // pv / pvT batch stride
  const long long strPQ = (long long)CNQ * CHA;
  const long long strAF = (long long)CNQ * CNV;  // aff / affT batch stride

  // ---- 2) vw = v @ Waff^T + baff         [B*Nv, H]
  launch_gemm(stream, CB * CNV, CH, CH, 1,
              vb, CH, 0, Waffb, CH, 0, vwb, CH, 0,
              baff, 1, 0, nullptr, 0, 0, nullptr, nullptr, 0);
  // ---- 3) pv = v @ Wv^T + bv             [B*Nv, Ha]
  launch_gemm(stream, CB * CNV, CHA, CH, 1,
              vb, CH, 0, Wvb, CH, 0, pvb, CHA, 0,
              bv, 1, 0, nullptr, 0, 0, nullptr, nullptr, 0);
  // ---- 4) pvT[b] = Wv @ v[b]^T + bv(m)   [Ha, Nv] per batch
  launch_gemm(stream, CHA, CNV, CH, CB,
              Wvb, CH, 0, vb, CH, strV, pvTb, CNV, strPV,
              bv, 2, 0, nullptr, 0, 0, nullptr, nullptr, 0);
  // ---- 5) pq = q @ Wq^T + bq             [B*Nq, Ha]
  launch_gemm(stream, CB * CNQ, CHA, CH, 1,
              qb, CH, 0, Wqb, CH, 0, pqb, CHA, 0,
              bq, 1, 0, nullptr, 0, 0, nullptr, nullptr, 0);
  // ---- 6) pqT[b] = Wq @ q[b]^T + bq(m)   [Ha, Nq] per batch
  launch_gemm(stream, CHA, CNQ, CH, CB,
              Wqb, CH, 0, qb, CH, strQ, pqTb, CNQ, strPQ,
              bq, 2, 0, nullptr, 0, 0, nullptr, nullptr, 0);
  // ---- 7) aff[b]  = tanh(q[b] @ vw[b]^T)   [Nq, Nv]
  launch_gemm(stream, CNQ, CNV, CH, CB,
              qb, CH, strQ, vwb, CH, strV, affb, CNV, strAF,
              nullptr, 0, 1, nullptr, 0, 0, nullptr, nullptr, 0);
  // ---- 8) affT[b] = tanh(vw[b] @ q[b]^T)   [Nv, Nq]
  launch_gemm(stream, CNV, CNQ, CH, CB,
              vwb, CH, strV, qb, CH, strQ, affTb, CNQ, strAF,
              nullptr, 0, 1, nullptr, 0, 0, nullptr, nullptr, 0);
  // ---- 9) sq[b,m] = tanh(pq + aff @ pv)[m,:] . Whq   (fused, h_q never stored)
  launch_gemm(stream, CNQ, CHA, CNV, CB,
              affb, CNV, strAF, pvTb, CNV, strPV, nullptr, 0, 0,
              nullptr, 0, 1, pqb, CHA, strPQ, Whq, sq, CNQ);
  // ---- 10) sv[b,m] = tanh(pv + aff^T @ pq)[m,:] . Whv (fused, h_v never stored)
  launch_gemm(stream, CNV, CHA, CNQ, CB,
              affTb, CNQ, strAF, pqTb, CNQ, strPQ, nullptr, 0, 0,
              nullptr, 0, 1, pvb, CHA, strPV, Whv, sv, CNV);

  // ---- 11) softmax over sequence + weighted sum of ORIGINAL fp32 v / q
  float* outv = (float*)d_out;            // v_hat [B, 1, H]
  float* outq = outv + (size_t)CB * CH;   // q_hat [B, 1, H]
  softmax_weighted_sum<<<dim3(CB), 256, 0, stream>>>(sv, v, outv, CNV);
  softmax_weighted_sum<<<dim3(CB), 256, 0, stream>>>(sq, q, outq, CNQ);
}
